// MultiHeadAttention_83726092468531
// MI455X (gfx1250) — matrix operand order repaired and checked by reading, not compiled, not run on hardware
//
#include <hip/hip_runtime.h>

// ---------------------------------------------------------------------------
// MI455X (gfx1250) multi-head prefix-LM attention, bf16 WMMA + f32 accumulate.
// B=8, S=512, D=1024, H=16, hd=64, NUM_PATCHES=196.
// GEMMs register-blocked 32x32 per wave: 1 b128 load per v_wmma.
// ---------------------------------------------------------------------------

typedef __bf16 bf16_t;
typedef bf16_t bfrag __attribute__((ext_vector_type(16))); // A/B fragment: 16 bf16 per lane
typedef float  f32x8 __attribute__((ext_vector_type(8)));  // C/D fragment: 8 f32 per lane

#define SEQ     512
#define DMODEL  1024
#define HEADS   16
#define HD      64
#define BATCH   8
#define NPATCH  196
#define NTOK    (BATCH * SEQ)   // 4096

union FragU { bfrag v; uint4 q[2]; };

__device__ __forceinline__ f32x8 zero8() {
  f32x8 z;
#pragma unroll
  for (int i = 0; i < 8; ++i) z[i] = 0.0f;
  return z;
}

// A-matrix fragment 16x32 bf16 from row-major memory (row stride in elements).
// ISA layout: lanes 0-15 -> row=lane,   K = {0..7, 16..23}
//             lanes16-31 -> row=lane-16, K = {8..15, 24..31}
__device__ __forceinline__ bfrag load_a_frag(const bf16_t* base, int row_stride, int lane) {
  const bf16_t* p = base + (size_t)(lane & 15) * row_stride + ((lane & 16) ? 8 : 0);
  FragU f;
  f.q[0] = *reinterpret_cast<const uint4*>(p);
  f.q[1] = *reinterpret_cast<const uint4*>(p + 16);
  return f.v;
}

// B-matrix fragment 32x16 bf16 from K-major memory (row stride in elements).
// ISA layout: lanes 0-15 -> col=lane, K = {0..7, 16..23}; lanes 16-31 -> col=lane-16, K = {8..15, 24..31}
__device__ __forceinline__ bfrag load_b_frag(const bf16_t* base, int row_stride, int lane) {
  const bf16_t* p = base + (lane & 15) + ((lane & 16) ? 8 : 0) * (size_t)row_stride;
  FragU f;
#pragma unroll
  for (int s = 0; s < 16; ++s) f.v[s] = p[(size_t)(s < 8 ? s : s + 8) * row_stride];
  return f.v;
}

__device__ __forceinline__ f32x8 wmma_bf16(bfrag a, bfrag b, f32x8 c) {
  return __builtin_amdgcn_wmma_f32_16x16x32_bf16(false, a, false, b, (short)0, c, false, false);
}

// ---------------------------------------------------------------------------
// Conversion kernels
// ---------------------------------------------------------------------------
__global__ void convert_x_kernel(const float* __restrict__ x, bf16_t* __restrict__ xb, int n) {
  for (int i = blockIdx.x * blockDim.x + threadIdx.x; i < n; i += gridDim.x * blockDim.x)
    xb[i] = (bf16_t)x[i];
}

// wt[k][m] = (bf16) w[m][k]  (K-major so it loads directly as a B fragment)
__global__ void transpose_w_kernel(const float* __restrict__ w, bf16_t* __restrict__ wt) {
  for (int i = blockIdx.x * blockDim.x + threadIdx.x; i < DMODEL * DMODEL;
       i += gridDim.x * blockDim.x) {
    int k = i >> 10, m = i & (DMODEL - 1);
    wt[i] = (bf16_t)w[m * DMODEL + k];
  }
}

// ---------------------------------------------------------------------------
// Fused QKV projection, 32x32 tile per wave (4 accumulators, 4 wmma / k-chunk).
// z=0: Q (pre-scaled by 1/sqrt(hd)=0.125, written transposed [b,h,hd,S])
// z=1: K  [b,h,S,hd]     z=2: V  [b,h,S,hd]
// ---------------------------------------------------------------------------
__global__ __launch_bounds__(128) void qkv_proj_kernel(
    const bf16_t* __restrict__ xb, const bf16_t* __restrict__ wtq,
    const bf16_t* __restrict__ wtk, const bf16_t* __restrict__ wtv,
    bf16_t* __restrict__ Qt, bf16_t* __restrict__ Kb, bf16_t* __restrict__ Vb) {
  const int lane = threadIdx.x & 31;
  const int wave = threadIdx.x >> 5;
  const int t0 = (blockIdx.x * 4 + wave) * 32;  // token tile (32 rows)
  const int m0 = blockIdx.y * 32;               // out-channel tile (32 cols)
  const int z  = blockIdx.z;
  const bf16_t* wt = (z == 0) ? wtq : (z == 1) ? wtk : wtv;

  f32x8 c[2][2];
#pragma unroll
  for (int i = 0; i < 2; ++i)
#pragma unroll
    for (int j = 0; j < 2; ++j) c[i][j] = zero8();

#pragma unroll 4
  for (int k0 = 0; k0 < DMODEL; k0 += 32) {
    bfrag a0 = load_a_frag(xb + (size_t)t0 * DMODEL + k0, DMODEL, lane);
    bfrag a1 = load_a_frag(xb + (size_t)(t0 + 16) * DMODEL + k0, DMODEL, lane);
    bfrag b0 = load_b_frag(wt + (size_t)k0 * DMODEL + m0, DMODEL, lane);
    bfrag b1 = load_b_frag(wt + (size_t)k0 * DMODEL + m0 + 16, DMODEL, lane);
    c[0][0] = wmma_bf16(a0, b0, c[0][0]);
    c[0][1] = wmma_bf16(a0, b1, c[0][1]);
    c[1][0] = wmma_bf16(a1, b0, c[1][0]);
    c[1][1] = wmma_bf16(a1, b1, c[1][1]);
  }

#pragma unroll
  for (int tt = 0; tt < 2; ++tt)
#pragma unroll
    for (int mt = 0; mt < 2; ++mt) {
      const int m = m0 + mt * 16 + (lane & 15);
      const int h = m >> 6, dh = m & (HD - 1);
#pragma unroll
      for (int j = 0; j < 8; ++j) {
        int t = t0 + tt * 16 + j + ((lane & 16) ? 8 : 0);
        int b = t >> 9, s = t & (SEQ - 1);
        float v = c[tt][mt][j];
        if (z == 0) {
          Qt[(((size_t)(b * HEADS + h)) * HD + dh) * SEQ + s] = (bf16_t)(v * 0.125f);
        } else if (z == 1) {
          Kb[(((size_t)(b * HEADS + h)) * SEQ + s) * HD + dh] = (bf16_t)v;
        } else {
          Vb[(((size_t)(b * HEADS + h)) * SEQ + s) * HD + dh] = (bf16_t)v;
        }
      }
    }
}

// ---------------------------------------------------------------------------
// Flash attention (prefix-LM mask). Scores computed transposed: S^T = K x Q^T,
// so each lane owns one query column -> softmax reduction is per-lane + one
// shfl_xor(16). P is bounced through per-wave LDS to re-fragment as A-matrix.
// Each wave handles 2 q-tiles (32 queries), streaming key chunks of 32.
// ---------------------------------------------------------------------------
__global__ __launch_bounds__(128) void flash_attn_kernel(
    const bf16_t* __restrict__ Qt, const bf16_t* __restrict__ Kb,
    const bf16_t* __restrict__ Vb, bf16_t* __restrict__ ctxb) {
  __shared__ bf16_t plds[4][16 * 32];  // per-wave P bounce buffer (1 KB each)
  const int lane = threadIdx.x & 31;
  const int wave = threadIdx.x >> 5;
  const int h = blockIdx.y, b = blockIdx.z;
  const int q_base = blockIdx.x * 128 + wave * 32;

  const bf16_t* Qh = Qt + (size_t)(b * HEADS + h) * HD * SEQ;   // [hd][S]
  const bf16_t* Kh = Kb + (size_t)(b * HEADS + h) * SEQ * HD;   // [S][hd]
  const bf16_t* Vh = Vb + (size_t)(b * HEADS + h) * SEQ * HD;   // [S][hd]
  bf16_t* plocal = plds[wave];

  // Q^T B-fragments: [q-tile][hd-chunk]; lane -> query column, 16 hd rows per lane.
  bfrag bq[2][2];
#pragma unroll
  for (int qt = 0; qt < 2; ++qt)
#pragma unroll
    for (int kc = 0; kc < 2; ++kc)
      bq[qt][kc] = load_b_frag(Qh + (size_t)(kc * 32) * SEQ + (q_base + qt * 16), SEQ, lane);

  f32x8 O[2][4];
  float mrow[2], lrow[2];
#pragma unroll
  for (int qt = 0; qt < 2; ++qt) {
    mrow[qt] = -1e30f;
    lrow[qt] = 0.0f;
#pragma unroll
    for (int dt = 0; dt < 4; ++dt) O[qt][dt] = zero8();
  }

  const int qlane = q_base + (lane & 15);  // query owned by this lane (tile 0)

  for (int sk0 = 0; sk0 < SEQ; sk0 += 32) {
    if (sk0 + 32 < SEQ) {  // speculative prefetch of next K/V chunk (global_prefetch_b8)
      __builtin_prefetch(Kh + (size_t)(sk0 + 32) * HD, 0, 1);
      __builtin_prefetch(Vh + (size_t)(sk0 + 32) * HD, 0, 1);
    }
    // K A-fragments: [sk-subtile][hd-chunk]
    bfrag aK[2][2];
#pragma unroll
    for (int t = 0; t < 2; ++t)
#pragma unroll
      for (int kc = 0; kc < 2; ++kc)
        aK[t][kc] = load_a_frag(Kh + (size_t)(sk0 + t * 16) * HD + kc * 32, HD, lane);
    // V B-fragments: [d-tile]; lane -> head-dim column, 16 key rows per lane.
    bfrag bV[4];
#pragma unroll
    for (int dt = 0; dt < 4; ++dt)
      bV[dt] = load_b_frag(Vh + (size_t)sk0 * HD + dt * 16, HD, lane);

#pragma unroll
    for (int qt = 0; qt < 2; ++qt) {
      // scores^T tiles (M=sk, N=q), f32 accumulate over hd=64
      f32x8 s[2];
#pragma unroll
      for (int t = 0; t < 2; ++t) {
        f32x8 c = zero8();
        c = wmma_bf16(aK[t][0], bq[qt][0], c);
        c = wmma_bf16(aK[t][1], bq[qt][1], c);
        s[t] = c;
      }
      const int q = qlane + qt * 16;
      // prefix-LM mask + per-lane partial row-max
      float pm = -1e30f;
#pragma unroll
      for (int t = 0; t < 2; ++t)
#pragma unroll
        for (int i = 0; i < 8; ++i) {
          int sk = sk0 + t * 16 + i + ((lane & 16) ? 8 : 0);
          bool ok = (q < NPATCH) ? (sk < NPATCH) : (sk <= q);
          float v = ok ? s[t][i] : -1e30f;
          s[t][i] = v;
          pm = fmaxf(pm, v);
        }
      float rm = fmaxf(pm, __shfl_xor(pm, 16, 32));
      float m_new = fmaxf(mrow[qt], rm);
      float ps = 0.0f;
#pragma unroll
      for (int t = 0; t < 2; ++t)
#pragma unroll
        for (int i = 0; i < 8; ++i) {
          float p = __expf(s[t][i] - m_new);
          s[t][i] = p;
          ps += p;
        }
      float rs = ps + __shfl_xor(ps, 16, 32);
      float corr = __expf(mrow[qt] - m_new);
      lrow[qt] = corr * lrow[qt] + rs;
      mrow[qt] = m_new;

      // Bounce P into LDS as [q_local][sk_local 0..31] (row-major, bf16).
      {
        bf16_t* pw = plocal + (lane & 15) * 32 + ((lane & 16) ? 8 : 0);
#pragma unroll
        for (int t = 0; t < 2; ++t)
#pragma unroll
          for (int i = 0; i < 8; ++i) pw[t * 16 + i] = (bf16_t)s[t][i];
      }
      // per-O-row correction factors (row q lives in lane q%16; bpermute broadcast)
      float c8[8];
#pragma unroll
      for (int j = 0; j < 8; ++j) c8[j] = __shfl(corr, j + ((lane & 16) ? 8 : 0), 32);
      // Re-fragment P as an A matrix straight out of LDS (ds_load_b128 pairs).
      bfrag aP = load_a_frag(plocal, 32, lane);
#pragma unroll
      for (int dt = 0; dt < 4; ++dt) {
        f32x8 o = O[qt][dt];
#pragma unroll
        for (int j = 0; j < 8; ++j) o[j] *= c8[j];
        O[qt][dt] = wmma_bf16(aP, bV[dt], o);
      }
    }
  }

  // Finalize: divide by row sum, write bf16 ctx[b][s][h*64+d]
#pragma unroll
  for (int qt = 0; qt < 2; ++qt) {
    float inv = 1.0f / lrow[qt];
    float i8[8];
#pragma unroll
    for (int j = 0; j < 8; ++j) i8[j] = __shfl(inv, j + ((lane & 16) ? 8 : 0), 32);
    const int d0 = lane & 15;
#pragma unroll
    for (int dt = 0; dt < 4; ++dt)
#pragma unroll
      for (int j = 0; j < 8; ++j) {
        int q = q_base + qt * 16 + j + ((lane & 16) ? 8 : 0);
        int d = h * HD + dt * 16 + d0;
        ctxb[((size_t)(b * SEQ + q)) * DMODEL + d] = (bf16_t)(O[qt][dt][j] * i8[j]);
      }
  }
}

// ---------------------------------------------------------------------------
// Output projection: out = ctx @ Wo^T + b_o (f32), 32x32 tile per wave.
// ---------------------------------------------------------------------------
__global__ __launch_bounds__(128) void out_proj_kernel(
    const bf16_t* __restrict__ ctxb, const bf16_t* __restrict__ wto,
    const float* __restrict__ bias, float* __restrict__ out) {
  const int lane = threadIdx.x & 31;
  const int wave = threadIdx.x >> 5;
  const int t0 = (blockIdx.x * 4 + wave) * 32;
  const int m0 = blockIdx.y * 32;

  f32x8 c[2][2];
#pragma unroll
  for (int i = 0; i < 2; ++i)
#pragma unroll
    for (int j = 0; j < 2; ++j) c[i][j] = zero8();

#pragma unroll 4
  for (int k0 = 0; k0 < DMODEL; k0 += 32) {
    bfrag a0 = load_a_frag(ctxb + (size_t)t0 * DMODEL + k0, DMODEL, lane);
    bfrag a1 = load_a_frag(ctxb + (size_t)(t0 + 16) * DMODEL + k0, DMODEL, lane);
    bfrag b0 = load_b_frag(wto + (size_t)k0 * DMODEL + m0, DMODEL, lane);
    bfrag b1 = load_b_frag(wto + (size_t)k0 * DMODEL + m0 + 16, DMODEL, lane);
    c[0][0] = wmma_bf16(a0, b0, c[0][0]);
    c[0][1] = wmma_bf16(a0, b1, c[0][1]);
    c[1][0] = wmma_bf16(a1, b0, c[1][0]);
    c[1][1] = wmma_bf16(a1, b1, c[1][1]);
  }

#pragma unroll
  for (int tt = 0; tt < 2; ++tt)
#pragma unroll
    for (int mt = 0; mt < 2; ++mt) {
      const int m = m0 + mt * 16 + (lane & 15);
      const float bo = bias[m];
#pragma unroll
      for (int j = 0; j < 8; ++j) {
        int t = t0 + tt * 16 + j + ((lane & 16) ? 8 : 0);
        out[(size_t)t * DMODEL + m] = c[tt][mt][j] + bo;
      }
    }
}

// ---------------------------------------------------------------------------
// Host-side launch
// ---------------------------------------------------------------------------
extern "C" void kernel_launch(void* const* d_in, const int* in_sizes, int n_in,
                              void* d_out, int out_size, void* d_ws, size_t ws_size,
                              hipStream_t stream) {
  (void)in_sizes; (void)n_in; (void)out_size; (void)ws_size;
  const float* x  = (const float*)d_in[0];
  const float* wq = (const float*)d_in[1];
  const float* wk = (const float*)d_in[2];
  const float* wv = (const float*)d_in[3];
  const float* wo = (const float*)d_in[4];
  const float* bo = (const float*)d_in[5];
  float* out = (float*)d_out;

  const size_t MB = 1ull << 20;
  char* ws = (char*)d_ws;
  bf16_t* xb   = (bf16_t*)(ws + 0 * MB);   // 8 MB  [4096][1024]
  bf16_t* wtq  = (bf16_t*)(ws + 8 * MB);   // 2 MB  [k][m]
  bf16_t* wtk  = (bf16_t*)(ws + 10 * MB);  // 2 MB
  bf16_t* wtv  = (bf16_t*)(ws + 12 * MB);  // 2 MB
  bf16_t* wto  = (bf16_t*)(ws + 14 * MB);  // 2 MB
  bf16_t* Qt   = (bf16_t*)(ws + 16 * MB);  // 8 MB  [b][h][hd][S]
  bf16_t* Kb   = (bf16_t*)(ws + 24 * MB);  // 8 MB  [b][h][S][hd]
  bf16_t* Vb   = (bf16_t*)(ws + 32 * MB);  // 8 MB  [b][h][S][hd]
  bf16_t* ctxb = (bf16_t*)(ws + 40 * MB);  // 8 MB  [b][S][D]

  convert_x_kernel<<<2048, 256, 0, stream>>>(x, xb, NTOK * DMODEL);
  transpose_w_kernel<<<1024, 256, 0, stream>>>(wq, wtq);
  transpose_w_kernel<<<1024, 256, 0, stream>>>(wk, wtk);
  transpose_w_kernel<<<1024, 256, 0, stream>>>(wv, wtv);
  transpose_w_kernel<<<1024, 256, 0, stream>>>(wo, wto);

  // token tiles: 4096/32 = 128 -> grid.x = 32 (4 waves/block along tokens)
  qkv_proj_kernel<<<dim3(32, 32, 3), 128, 0, stream>>>(xb, wtq, wtk, wtv, Qt, Kb, Vb);
  flash_attn_kernel<<<dim3(4, HEADS, BATCH), 128, 0, stream>>>(Qt, Kb, Vb, ctxb);
  out_proj_kernel<<<dim3(32, 32), 128, 0, stream>>>(ctxb, wto, bo, out);
}
